// JointAttention_89893665506036
// MI455X (gfx1250) — matrix operand order repaired and checked by reading, not compiled, not run on hardware
//
#include <hip/hip_runtime.h>

// ---------- types ----------
typedef __attribute__((ext_vector_type(16))) __bf16 v16bf;
typedef __attribute__((ext_vector_type(8)))  float  v8f;
typedef __attribute__((ext_vector_type(4)))  unsigned int uint4v;
typedef __attribute__((ext_vector_type(4)))  int    int4v;
typedef __attribute__((ext_vector_type(4)))  float  float4v;
typedef __attribute__((ext_vector_type(2)))  float  float2v;

// address-space qualified vector pointees for the async-LDS builtin
typedef __attribute__((address_space(1))) int4v gas_int4v;   // global
typedef __attribute__((address_space(3))) int4v las_int4v;   // LDS

union FragBF { uint4v u[2]; v16bf v; };

// ---------- feature probes (compile-safe) ----------
#if defined(__has_builtin)
#  if __has_builtin(__builtin_amdgcn_global_load_async_to_lds_b128)
#    define HAVE_ASYNC_LDS 1
#  endif
#  if __has_builtin(__builtin_amdgcn_cvt_pk_bf16_f32)
#    define HAVE_PK_BF16 1
#  endif
#  if __has_builtin(__builtin_amdgcn_s_wait_asynccnt)
#    define HAVE_WAIT_ASYNC 1
#  endif
#endif
#ifndef HAVE_ASYNC_LDS
#  define HAVE_ASYNC_LDS 0
#endif
#ifndef HAVE_PK_BF16
#  define HAVE_PK_BF16 0
#endif
#ifndef HAVE_WAIT_ASYNC
#  define HAVE_WAIT_ASYNC 0
#endif

__device__ __forceinline__ unsigned short f2b(float f) {
    union { float f; unsigned u; } x; x.f = f;
    unsigned r = x.u + 0x7FFFu + ((x.u >> 16) & 1u);   // round-to-nearest-even
    return (unsigned short)(r >> 16);
}

// packed fp32x2 -> bf16x2 (single v_cvt_pk_bf16_f32 when available)
__device__ __forceinline__ unsigned pkbf(float x, float y) {
#if HAVE_PK_BF16
    auto r = __builtin_amdgcn_cvt_pk_bf16_f32(x, y);
    unsigned u; __builtin_memcpy(&u, &r, 4);
    return u;
#else
    return (unsigned)f2b(x) | ((unsigned)f2b(y) << 16);
#endif
}

// async global -> LDS 16-byte copy (ASYNCcnt path on gfx1250)
__device__ __forceinline__ void async_copy16(void* lds, const void* g) {
#if HAVE_ASYNC_LDS
    __builtin_amdgcn_global_load_async_to_lds_b128(
        (gas_int4v*)g, (las_int4v*)lds, 0, 0);
#else
    *(uint4v*)lds = *(const uint4v*)g;
#endif
}

__device__ __forceinline__ void wait_async() {
#if HAVE_ASYNC_LDS
#  if HAVE_WAIT_ASYNC
    __builtin_amdgcn_s_wait_asynccnt(0);
#  else
    asm volatile("s_wait_asynccnt 0" ::: "memory");
#  endif
#endif
}

__device__ __forceinline__ v8f wmma_bf16(v16bf a, v16bf b, v8f c) {
    return __builtin_amdgcn_wmma_f32_16x16x32_bf16(
        /*neg_a=*/false, a, /*neg_b=*/false, b,
        /*c_mod=*/(short)0, c, /*reuse_a=*/false, /*reuse_b=*/false);
}

// ------------------------------------------------------------------
// Generic GEMM: C[M][N] = A[M][K] (fp32) * W[K][N] (fp32) + bias[N]
// bf16 WMMA, fp32 accumulate. Block tile 128(M) x 64(N), BK=32,
// 256 threads = 8 waves, each wave: 16 rows x 64 cols (4 C fragments).
// ------------------------------------------------------------------
__global__ __launch_bounds__(256) void gemm_bf16_wmma(
    const float* __restrict__ A, const float* __restrict__ W,
    const float* __restrict__ bias, float* __restrict__ C,
    int M, int N, int K)
{
    __shared__ __align__(16) unsigned short lds_a[128 * 40]; // 128 rows x 32 (+8 pad) bf16
    __shared__ __align__(16) unsigned short lds_b[64 * 40];  // 64 N-rows x 32 K (transposed)

    const int tid  = threadIdx.x;
    const int lane = tid & 31;
    const int wv   = tid >> 5;
    const int lm   = lane & 15;
    const int half = lane >> 4;
    const int m0   = blockIdx.y * 128;
    const int n0   = blockIdx.x * 64;

    v8f acc0 = {}; v8f acc1 = {}; v8f acc2 = {}; v8f acc3 = {};

    for (int kt = 0; kt < K; kt += 32) {
        // prefetch next K-step tiles into L2/WGP$ (global_prefetch_b8)
        if (kt + 32 < K) {
            __builtin_prefetch(A + (size_t)(m0 + (tid >> 1)) * K + kt + 32 + (tid & 1) * 16, 0, 1);
            __builtin_prefetch(W + (size_t)(kt + 32 + (tid >> 3)) * N + n0 + (tid & 7) * 8, 0, 1);
        }
        // Stage A tile: 128x32 fp32 -> bf16 LDS (row-major, padded), packed stores
        #pragma unroll
        for (int i = 0; i < 4; ++i) {
            int lin = tid + i * 256;              // 0..1023
            int row = lin >> 3;
            int c4  = (lin & 7) << 2;
            float4v f = *(const float4v*)(A + (size_t)(m0 + row) * K + kt + c4);
            unsigned* p = (unsigned*)&lds_a[row * 40 + c4];
            p[0] = pkbf(f.x, f.y);
            p[1] = pkbf(f.z, f.w);
        }
        // Stage B tile: 32(K)x64(N) fp32 -> transposed bf16 LDS [n][k]
        #pragma unroll
        for (int i = 0; i < 2; ++i) {
            int lin = tid + i * 256;              // 0..511
            int kr  = lin >> 4;                   // 0..31
            int n4  = (lin & 15) << 2;
            float4v f = *(const float4v*)(W + (size_t)(kt + kr) * N + n0 + n4);
            lds_b[(n4 + 0) * 40 + kr] = f2b(f.x);
            lds_b[(n4 + 1) * 40 + kr] = f2b(f.y);
            lds_b[(n4 + 2) * 40 + kr] = f2b(f.z);
            lds_b[(n4 + 3) * 40 + kr] = f2b(f.w);
        }
        __syncthreads();

        // A fragment: lane holds row (wv*16+lm); e0..15 -> K=half*16+0..15
        FragBF af;
        af.u[0] = *(const uint4v*)&lds_a[(wv * 16 + lm) * 40 + half * 16];
        af.u[1] = *(const uint4v*)&lds_a[(wv * 16 + lm) * 40 + half * 16 + 8];
        // B fragments: lane holds column (nf*16+lm), K = half*16 + e (contiguous in lds_b row)
        FragBF b0, b1, b2, b3;
        b0.u[0] = *(const uint4v*)&lds_b[( 0 + lm) * 40 + half * 16];
        b0.u[1] = *(const uint4v*)&lds_b[( 0 + lm) * 40 + half * 16 + 8];
        b1.u[0] = *(const uint4v*)&lds_b[(16 + lm) * 40 + half * 16];
        b1.u[1] = *(const uint4v*)&lds_b[(16 + lm) * 40 + half * 16 + 8];
        b2.u[0] = *(const uint4v*)&lds_b[(32 + lm) * 40 + half * 16];
        b2.u[1] = *(const uint4v*)&lds_b[(32 + lm) * 40 + half * 16 + 8];
        b3.u[0] = *(const uint4v*)&lds_b[(48 + lm) * 40 + half * 16];
        b3.u[1] = *(const uint4v*)&lds_b[(48 + lm) * 40 + half * 16 + 8];

        acc0 = wmma_bf16(af.v, b0.v, acc0);
        acc1 = wmma_bf16(af.v, b1.v, acc1);
        acc2 = wmma_bf16(af.v, b2.v, acc2);
        acc3 = wmma_bf16(af.v, b3.v, acc3);
        __syncthreads();
    }

    const float bv0 = bias[n0 +  0 + lm];
    const float bv1 = bias[n0 + 16 + lm];
    const float bv2 = bias[n0 + 32 + lm];
    const float bv3 = bias[n0 + 48 + lm];
    #pragma unroll
    for (int r = 0; r < 8; ++r) {
        size_t row = (size_t)(m0 + wv * 16 + r + 8 * half) * N + n0;
        C[row +  0 + lm] = acc0[r] + bv0;
        C[row + 16 + lm] = acc1[r] + bv1;
        C[row + 32 + lm] = acc2[r] + bv2;
        C[row + 48 + lm] = acc3[r] + bv3;
    }
}

// ------------------------------------------------------------------
// RMSNorm(q,k) per head + bf16 conversion + layout:
//   Q,K -> [B][H][Ntot][64] bf16 ; V -> [B][H][64][Ntot] bf16 (transposed)
// qkv fp32 rows: [B*rows][3*1024] with [3][16][64] column layout.
// One wave per (b, n, h) task; 8 tasks per 256-thread block.
// ------------------------------------------------------------------
__global__ __launch_bounds__(256) void norm_layout(
    const float* __restrict__ qkv, const float* __restrict__ sq,
    const float* __restrict__ sk, unsigned short* __restrict__ Qo,
    unsigned short* __restrict__ Ko, unsigned short* __restrict__ Vt,
    int rows, int nOffset)
{
    const int lane = threadIdx.x & 31;
    const int wv   = threadIdx.x >> 5;
    const int task = blockIdx.x * 8 + wv;
    const int b  = task / (rows * 16);
    const int t2 = task - b * rows * 16;
    const int n  = t2 >> 4;
    const int h  = t2 & 15;

    const float* src = qkv + (size_t)(b * rows + n) * 3072;
    float2v qv = *(const float2v*)(src + h * 64 + 2 * lane);
    float2v kv = *(const float2v*)(src + 1024 + h * 64 + 2 * lane);
    float2v vv = *(const float2v*)(src + 2048 + h * 64 + 2 * lane);

    float sumq = qv.x * qv.x + qv.y * qv.y;
    float sumk = kv.x * kv.x + kv.y * kv.y;
    #pragma unroll
    for (int m = 1; m < 32; m <<= 1) {
        sumq += __shfl_xor(sumq, m);
        sumk += __shfl_xor(sumk, m);
    }
    const float rq = rsqrtf(sumq * (1.0f / 64.0f) + 1e-6f);
    const float rk = rsqrtf(sumk * (1.0f / 64.0f) + 1e-6f);
    float2v sqv = *(const float2v*)(sq + 2 * lane);
    float2v skv = *(const float2v*)(sk + 2 * lane);

    unsigned qo = pkbf(sqv.x * qv.x * rq, sqv.y * qv.y * rq);
    unsigned ko = pkbf(skv.x * kv.x * rk, skv.y * kv.y * rk);
    size_t base = ((size_t)(b * 16 + h) * 2304 + nOffset + n) * 64 + 2 * lane;
    *(unsigned*)(Qo + base) = qo;
    *(unsigned*)(Ko + base) = ko;

    size_t vb = ((size_t)(b * 16 + h) * 64 + 2 * lane) * 2304 + nOffset + n;
    Vt[vb]        = f2b(vv.x);
    Vt[vb + 2304] = f2b(vv.y);
}

// ------------------------------------------------------------------
// Flash attention (full, no mask). Grid (qtile, head, batch), 128 thr.
// Each wave: 16 queries, full d=64, online softmax over 2304 keys in
// 32-key blocks. K/V tiles are staged once per block into LDS via
// async global->LDS copies (shared by all 4 waves), then WMMA
// fragments are read from LDS. 8 WMMAs per key block per wave.
// ------------------------------------------------------------------
__global__ __launch_bounds__(128) void flash_attn(
    const unsigned short* __restrict__ Q, const unsigned short* __restrict__ Km,
    const unsigned short* __restrict__ Vt, float* __restrict__ Out)
{
    const int Ntot = 2304, H = 16, D = 64;
    const int lane = threadIdx.x & 31;
    const int wv   = threadIdx.x >> 5;
    const int lm   = lane & 15;
    const int half = lane >> 4;
    const int qt = blockIdx.x, h = blockIdx.y, b = blockIdx.z;
    const size_t bh = (size_t)(b * H + h);
    const unsigned short* Qh = Q  + bh * Ntot * D;
    const unsigned short* Kh = Km + bh * Ntot * D;
    const unsigned short* Vh = Vt + bh * D * Ntot;
    const int q0 = qt * 64 + wv * 16;

    __shared__ __align__(16) unsigned short Kt[32 * 64];   // [key][d]   4KB
    __shared__ __align__(16) unsigned short Vl[64 * 32];   // [d][key]   4KB
    __shared__ __align__(16) unsigned short Pl[4][16 * 32];
    unsigned short* P = Pl[wv];

    // Q A-fragments for the two 32-wide d chunks (held in VGPRs all along)
    FragBF qf0, qf1;
    {
        const unsigned short* qp = Qh + (size_t)(q0 + lm) * D;
        qf0.u[0] = *(const uint4v*)(qp + half * 16);
        qf0.u[1] = *(const uint4v*)(qp + half * 16 + 8);
        qf1.u[0] = *(const uint4v*)(qp + 32 + half * 16);
        qf1.u[1] = *(const uint4v*)(qp + 32 + half * 16 + 8);
    }

    v8f o0 = {}, o1 = {}, o2 = {}, o3 = {};
    float mrow[8], lrow[8];
    #pragma unroll
    for (int r = 0; r < 8; ++r) { mrow[r] = -1e30f; lrow[r] = 0.0f; }
    const float scale = 0.125f; // 1/sqrt(64)

    for (int kb = 0; kb < Ntot; kb += 32) {
        // ---- async-stage K tile (32 keys x 64 d) and V^T tile (64 d x 32 keys)
        // 256 16B chunks each; 128 threads x 2 chunks.
        #pragma unroll
        for (int i = 0; i < 2; ++i) {
            int ck = threadIdx.x * 2 + i;          // 0..255
            // K: chunk ck -> key = ck/8, 8 ushorts at offset (ck%8)*8
            async_copy16(&Kt[ck * 8],
                         Kh + (size_t)(kb + (ck >> 3)) * D + (ck & 7) * 8);
            // V: chunk ck -> d = ck/4, 8 ushorts at offset (ck%4)*8
            int d = ck >> 2, vo = (ck & 3) * 8;
            async_copy16(&Vl[d * 32 + vo],
                         Vh + (size_t)d * Ntot + kb + vo);
        }
        wait_async();
        __syncthreads();

        // ---- scores S[16q x 32k]: d contracted in two K=32 WMMAs per 16-key tile
        v8f s0 = {}, s1 = {};
        {
            FragBF kf;
            const unsigned short* kp = &Kt[lm * 64];            // keys kb..kb+15
            kf.u[0] = *(const uint4v*)(kp + half * 16);
            kf.u[1] = *(const uint4v*)(kp + half * 16 + 8);
            s0 = wmma_bf16(qf0.v, kf.v, s0);
            kf.u[0] = *(const uint4v*)(kp + 32 + half * 16);
            kf.u[1] = *(const uint4v*)(kp + 32 + half * 16 + 8);
            s0 = wmma_bf16(qf1.v, kf.v, s0);
            const unsigned short* kp2 = &Kt[(16 + lm) * 64];    // keys kb+16..kb+31
            kf.u[0] = *(const uint4v*)(kp2 + half * 16);
            kf.u[1] = *(const uint4v*)(kp2 + half * 16 + 8);
            s1 = wmma_bf16(qf0.v, kf.v, s1);
            kf.u[0] = *(const uint4v*)(kp2 + 32 + half * 16);
            kf.u[1] = *(const uint4v*)(kp2 + 32 + half * 16 + 8);
            s1 = wmma_bf16(qf1.v, kf.v, s1);
        }
        // ---- online softmax (row reductions inside each 16-lane n-group)
        #pragma unroll
        for (int r = 0; r < 8; ++r) {
            float v0 = s0[r] * scale, v1 = s1[r] * scale;
            float bm = fmaxf(v0, v1);
            bm = fmaxf(bm, __shfl_xor(bm, 1));
            bm = fmaxf(bm, __shfl_xor(bm, 2));
            bm = fmaxf(bm, __shfl_xor(bm, 4));
            bm = fmaxf(bm, __shfl_xor(bm, 8));
            float mnew = fmaxf(mrow[r], bm);
            float corr = __expf(mrow[r] - mnew);
            float p0 = __expf(v0 - mnew);
            float p1 = __expf(v1 - mnew);
            float ps = p0 + p1;
            ps += __shfl_xor(ps, 1);
            ps += __shfl_xor(ps, 2);
            ps += __shfl_xor(ps, 4);
            ps += __shfl_xor(ps, 8);
            lrow[r] = lrow[r] * corr + ps;
            mrow[r] = mnew;
            o0[r] *= corr; o1[r] *= corr; o2[r] *= corr; o3[r] *= corr;
            P[(r + 8 * half) * 32 + lm]      = f2b(p0);
            P[(r + 8 * half) * 32 + 16 + lm] = f2b(p1);
        }
        // ---- P (16x32) as A-fragment from per-wave LDS (in-order DS, same wave)
        FragBF pf;
        pf.u[0] = *(const uint4v*)&P[lm * 32 + half * 16];
        pf.u[1] = *(const uint4v*)&P[lm * 32 + half * 16 + 8];
        // ---- PV: V^T in LDS [d][key]; lane's column d = g*16+lm contiguous over keys
        {
            FragBF vf;
            const unsigned short* vp = &Vl[lm * 32 + half * 16];
            vf.u[0] = *(const uint4v*)(vp);
            vf.u[1] = *(const uint4v*)(vp + 8);
            o0 = wmma_bf16(pf.v, vf.v, o0);
            vf.u[0] = *(const uint4v*)(vp + 16 * 32);
            vf.u[1] = *(const uint4v*)(vp + 16 * 32 + 8);
            o1 = wmma_bf16(pf.v, vf.v, o1);
            vf.u[0] = *(const uint4v*)(vp + 32 * 32);
            vf.u[1] = *(const uint4v*)(vp + 32 * 32 + 8);
            o2 = wmma_bf16(pf.v, vf.v, o2);
            vf.u[0] = *(const uint4v*)(vp + 48 * 32);
            vf.u[1] = *(const uint4v*)(vp + 48 * 32 + 8);
            o3 = wmma_bf16(pf.v, vf.v, o3);
        }
        __syncthreads();   // protect K/V tiles before next async stage
    }

    #pragma unroll
    for (int r = 0; r < 8; ++r) {
        float inv = 1.0f / lrow[r];
        size_t row = ((size_t)b * Ntot + q0 + r + 8 * half) * 1024 + h * 64;
        Out[row +  0 + lm] = o0[r] * inv;
        Out[row + 16 + lm] = o1[r] * inv;
        Out[row + 32 + lm] = o2[r] * inv;
        Out[row + 48 + lm] = o3[r] * inv;
    }
}

// ------------------------------------------------------------------
extern "C" void kernel_launch(void* const* d_in, const int* in_sizes, int n_in,
                              void* d_out, int out_size, void* d_ws, size_t ws_size,
                              hipStream_t stream) {
    (void)in_sizes; (void)n_in; (void)out_size; (void)ws_size;
    const float* x        = (const float*)d_in[0];
    const float* c        = (const float*)d_in[1];
    const float* w_qkv_x  = (const float*)d_in[2];
    const float* b_qkv_x  = (const float*)d_in[3];
    const float* w_qkv_c  = (const float*)d_in[4];
    const float* b_qkv_c  = (const float*)d_in[5];
    const float* s_q_x    = (const float*)d_in[6];
    const float* s_k_x    = (const float*)d_in[7];
    const float* s_q_c    = (const float*)d_in[8];
    const float* s_k_c    = (const float*)d_in[9];
    const float* w_proj_x = (const float*)d_in[10];
    const float* b_proj_x = (const float*)d_in[11];
    const float* w_proj_c = (const float*)d_in[12];
    const float* b_proj_c = (const float*)d_in[13];
    float* out = (float*)d_out;

    // workspace carve (all chunks 16B-aligned)
    char* wsp = (char*)d_ws;
    float* qkv_x = (float*)wsp;           wsp += (size_t)4096 * 3072 * sizeof(float);
    float* qkv_c = (float*)wsp;           wsp += (size_t)512  * 3072 * sizeof(float);
    unsigned short* Qb = (unsigned short*)wsp; wsp += (size_t)2 * 16 * 2304 * 64 * 2;
    unsigned short* Kb = (unsigned short*)wsp; wsp += (size_t)2 * 16 * 2304 * 64 * 2;
    unsigned short* Vt = (unsigned short*)wsp; wsp += (size_t)2 * 16 * 2304 * 64 * 2;
    float* aout = (float*)wsp;            wsp += (size_t)2 * 2304 * 1024 * sizeof(float);

    // 1) QKV GEMMs (fp32 in, bf16 WMMA, fp32 out)
    gemm_bf16_wmma<<<dim3(3072 / 64, 4096 / 128), 256, 0, stream>>>(
        x, w_qkv_x, b_qkv_x, qkv_x, 4096, 3072, 1024);
    gemm_bf16_wmma<<<dim3(3072 / 64, 512 / 128), 256, 0, stream>>>(
        c, w_qkv_c, b_qkv_c, qkv_c, 512, 3072, 1024);

    // 2) RMSNorm + bf16 layout (x part -> tokens [0,2048), c part -> [2048,2304))
    norm_layout<<<(2 * 2048 * 16) / 8, 256, 0, stream>>>(
        qkv_x, s_q_x, s_k_x, Qb, Kb, Vt, 2048, 0);
    norm_layout<<<(2 * 256 * 16) / 8, 256, 0, stream>>>(
        qkv_c, s_q_c, s_k_c, Qb, Kb, Vt, 256, 2048);

    // 3) Flash attention: 36 q-tiles x 16 heads x 2 batches
    flash_attn<<<dim3(2304 / 64, 16, 2), 128, 0, stream>>>(Qb, Kb, Vt, aout);

    // 4) Output projections (per batch, per segment)
    for (int bb = 0; bb < 2; ++bb) {
        gemm_bf16_wmma<<<dim3(1024 / 64, 2048 / 128), 256, 0, stream>>>(
            aout + (size_t)bb * 2304 * 1024, w_proj_x, b_proj_x,
            out + (size_t)bb * 2048 * 1024, 2048, 1024, 1024);
        gemm_bf16_wmma<<<dim3(1024 / 64, 256 / 128), 256, 0, stream>>>(
            aout + ((size_t)bb * 2304 + 2048) * 1024, w_proj_c, b_proj_c,
            out + (size_t)(2 * 2048 * 1024) + (size_t)bb * 256 * 1024, 256, 1024, 1024);
    }
}
